// CustomAttention_34333968564372
// MI455X (gfx1250) — matrix operand order repaired and checked by reading, not compiled, not run on hardware
//
#include <hip/hip_runtime.h>

// ---------------------------------------------------------------------------
// CDNA5 (gfx1250, wave32) sliding-window attention.
//   K1: per-token projections (f32 VALU) -> f16 Q/K (d padded 24->32, scale
//       1/sqrt(24) folded into Q) and f16 V stored transposed Vt[d][l].
//   K2: one wave per (b,h,16-query tile). 8x v_wmma_f32_16x16x32_f16 for QK^T
//       (16x128 scores), in-register softmax (shfl_xor over 16-lane halves),
//       LDS transpose of P to A-layout, 8x WMMA for P.V, late 1/rowsum.
//       Mask uses one unsigned range-compare per (row,tile).
//   K3: head merge + output projections (f32 VALU).
// ---------------------------------------------------------------------------

typedef __attribute__((ext_vector_type(16))) _Float16 v16h;
typedef __attribute__((ext_vector_type(8)))  _Float16 v8h;
typedef __attribute__((ext_vector_type(8)))  float    v8f;

#define Bsz 8
#define Lsz 2048
#define Hn  4
#define Dh  24     // logical head dim (6 ang_r + 6 ang_i + 12 amp)
#define Dp  32     // padded head dim for WMMA K=32
#define WIN 100

union ABfrag { v16h v; v8h h[2]; };

// ------------------------------ Kernel 1 -----------------------------------
__global__ __launch_bounds__(256) void proj_kernel(
    const float* __restrict__ x_ang, const float* __restrict__ x_amp,
    const float* __restrict__ Wq_amp, const float* __restrict__ Wk_amp,
    const float* __restrict__ Wv_amp,
    const float* __restrict__ Wq_r, const float* __restrict__ Wq_i,
    const float* __restrict__ Wk_r, const float* __restrict__ Wk_i,
    const float* __restrict__ Wv_r, const float* __restrict__ Wv_i,
    _Float16* __restrict__ Q, _Float16* __restrict__ K,
    _Float16* __restrict__ Vt)
{
  int idx = blockIdx.x * blockDim.x + threadIdx.x;   // b*L + l
  if (idx >= Bsz * Lsz) return;
  int b = idx / Lsz, l = idx % Lsz;
  const float* xa = x_amp + (size_t)idx * 48;
  const float* xr = x_ang + (size_t)idx * 32;   // first 16 = real
  const float* xi = xr + 16;                    // next 16  = imag

  float qamp[48], kamp[48], vamp[48];
  for (int o = 0; o < 48; ++o) {
    float aq = 0.f, ak = 0.f, av = 0.f;
    for (int i = 0; i < 48; ++i) {
      float x = xa[i];
      aq += x * Wq_amp[i * 48 + o];
      ak += x * Wk_amp[i * 48 + o];
      av += x * Wv_amp[i * 48 + o];
    }
    qamp[o] = aq; kamp[o] = ak; vamp[o] = av;
  }

  float qar[24], qai[24], kar[24], kai[24], var_[24], vai[24];
  for (int o = 0; o < 24; ++o) {
    float qr = 0.f, qi2 = 0.f, kr = 0.f, ki = 0.f, vr = 0.f, vi = 0.f;
    for (int i = 0; i < 16; ++i) {
      float r = xr[i], im = xi[i];
      float wqr = Wq_r[i * 24 + o], wqi = Wq_i[i * 24 + o];
      float wkr = Wk_r[i * 24 + o], wki = Wk_i[i * 24 + o];
      float wvr = Wv_r[i * 24 + o], wvi = Wv_i[i * 24 + o];
      qr  += r * wqr - im * wqi;  qi2 += r * wqi + im * wqr;
      kr  += r * wkr - im * wki;  ki  += r * wki + im * wkr;
      vr  += r * wvr - im * wvi;  vi  += r * wvi + im * wvr;
    }
    qar[o] = qr; qai[o] = qi2; kar[o] = kr; kai[o] = ki;
    var_[o] = vr; vai[o] = vi;
  }

  const float scale = 0.20412414523193154f;  // 1/sqrt(24), folded into Q
  for (int h = 0; h < Hn; ++h) {
    size_t bh = (size_t)(b * Hn + h);
    _Float16* qrow = Q + (bh * Lsz + l) * Dp;
    _Float16* krow = K + (bh * Lsz + l) * Dp;
    for (int j = 0; j < 6; ++j) {
      qrow[j]     = (_Float16)(qar[h * 6 + j] * scale);
      qrow[6 + j] = (_Float16)(qai[h * 6 + j] * scale);
      krow[j]     = (_Float16)(kar[h * 6 + j]);
      krow[6 + j] = (_Float16)(kai[h * 6 + j]);
    }
    for (int j = 0; j < 12; ++j) {
      qrow[12 + j] = (_Float16)(qamp[h * 12 + j] * scale);
      krow[12 + j] = (_Float16)(kamp[h * 12 + j]);
    }
    for (int j = 24; j < 32; ++j) { qrow[j] = (_Float16)0.f; krow[j] = (_Float16)0.f; }

    // V stored transposed: Vt[bh][d][l], d padded to 32 rows (zeros 24..31)
    _Float16* vcol = Vt + (bh * Dp) * Lsz + l;
    for (int j = 0; j < 6; ++j) {
      vcol[(size_t)j * Lsz]       = (_Float16)var_[h * 6 + j];
      vcol[(size_t)(6 + j) * Lsz] = (_Float16)vai[h * 6 + j];
    }
    for (int j = 0; j < 12; ++j)
      vcol[(size_t)(12 + j) * Lsz] = (_Float16)vamp[h * 12 + j];
    for (int d = 24; d < 32; ++d)
      vcol[(size_t)d * Lsz] = (_Float16)0.f;
  }
}

// ------------------------------ Kernel 2 -----------------------------------
__global__ __launch_bounds__(256) void attn_kernel(
    const _Float16* __restrict__ Q, const _Float16* __restrict__ K,
    const _Float16* __restrict__ Vt, float* __restrict__ O)
{
  __shared__ _Float16 plds[8][16 * 128];   // per-wave P transpose slab (4KB)

  int wave = threadIdx.x >> 5;
  int lane = threadIdx.x & 31;
  int gw   = blockIdx.x * 8 + wave;        // 4096 waves total
  int qt   = gw & 127;                     // query tile within (b,h)
  int bh   = gw >> 7;                      // b*H + h
  int q0   = qt * 16;
  int kstart = q0 - 112; if (kstart < 0) kstart = 0;  // 8 key tiles cover window

  const _Float16* Qbh = Q  + (size_t)bh * Lsz * Dp;
  const _Float16* Kbh = K  + (size_t)bh * Lsz * Dp;
  const _Float16* Vbh = Vt + (size_t)bh * Dp * Lsz;

  int half = lane >> 4;                    // 16-lane half-group
  int n    = lane & 15;
  int kb0  = half ? 16 : 0;                // A-matrix K packing: contiguous k = 16*half + s
  int kb1  = half ? 24 : 8;
  int kbB  = half ? 16 : 0;                // B-matrix K base for this half

  __builtin_prefetch(Kbh + (size_t)(kstart + n) * Dp, 0, 3);
  __builtin_prefetch(Vbh + (size_t)n * Lsz + kstart, 0, 3);

  // A fragment: Q tile 16x32, lane holds row m=n
  ABfrag aQ;
  aQ.h[0] = *(const v8h*)(Qbh + (size_t)(q0 + n) * Dp + kb0);
  aQ.h[1] = *(const v8h*)(Qbh + (size_t)(q0 + n) * Dp + kb1);

  // scores: 16x128 across 8 WMMA accumulators
  v8f s[8];
  v8f cz = {};
  ABfrag bK;
  #pragma unroll
  for (int t = 0; t < 8; ++t) {
    const _Float16* kp = Kbh + (size_t)(kstart + t * 16 + n) * Dp + kbB;
    bK.h[0] = *(const v8h*)(kp);
    bK.h[1] = *(const v8h*)(kp + 8);
    s[t] = __builtin_amdgcn_wmma_f32_16x16x32_f16(
        false, aQ.v, false, bK.v, (short)0, cz, false, false);
  }

  // mask + softmax (numerator only; normalize after PV).
  // valid  <=>  0 <= q-k < WIN  : single unsigned range compare.
  float rowsum[8];
  #pragma unroll
  for (int r = 0; r < 8; ++r) {
    int m = r + half * 8;
    int q = q0 + m;
    float mx = -1e30f;
    #pragma unroll
    for (int t = 0; t < 8; ++t) {
      int k = kstart + t * 16 + n;
      bool valid = (unsigned)(q - k) < (unsigned)WIN;
      float sv = valid ? s[t][r] : -1e30f;
      s[t][r] = sv;
      mx = fmaxf(mx, sv);
    }
    mx = fmaxf(mx, __shfl_xor(mx, 1));
    mx = fmaxf(mx, __shfl_xor(mx, 2));
    mx = fmaxf(mx, __shfl_xor(mx, 4));
    mx = fmaxf(mx, __shfl_xor(mx, 8));
    float sum = 0.f;
    #pragma unroll
    for (int t = 0; t < 8; ++t) {
      float p = __expf(s[t][r] - mx);
      s[t][r] = p;
      sum += p;
    }
    sum += __shfl_xor(sum, 1);
    sum += __shfl_xor(sum, 2);
    sum += __shfl_xor(sum, 4);
    sum += __shfl_xor(sum, 8);
    rowsum[r] = sum;
  }

  // C-layout -> A-layout transpose of P through LDS
  _Float16* pl = plds[wave];
  #pragma unroll
  for (int r = 0; r < 8; ++r) {
    int m = r + half * 8;
    #pragma unroll
    for (int t = 0; t < 8; ++t)
      pl[m * 128 + t * 16 + n] = (_Float16)s[t][r];
  }

  // P (16x128) @ V (128x24, padded 32): 4 K-chunks x 2 N-tiles
  v8f acc0 = {}, acc1 = {};
  ABfrag aP, bV0, bV1;
  #pragma unroll
  for (int c = 0; c < 4; ++c) {
    aP.h[0] = *(const v8h*)(pl + n * 128 + c * 32 + kb0);
    aP.h[1] = *(const v8h*)(pl + n * 128 + c * 32 + kb1);
    const _Float16* v0 = Vbh + (size_t)n        * Lsz + kstart + c * 32 + kbB;
    const _Float16* v1 = Vbh + (size_t)(16 + n) * Lsz + kstart + c * 32 + kbB;
    bV0.h[0] = *(const v8h*)(v0); bV0.h[1] = *(const v8h*)(v0 + 8);
    bV1.h[0] = *(const v8h*)(v1); bV1.h[1] = *(const v8h*)(v1 + 8);
    acc0 = __builtin_amdgcn_wmma_f32_16x16x32_f16(
        false, aP.v, false, bV0.v, (short)0, acc0, false, false);
    acc1 = __builtin_amdgcn_wmma_f32_16x16x32_f16(
        false, aP.v, false, bV1.v, (short)0, acc1, false, false);
  }

  // normalize + store O[bh][q][d] (d<24)
  float* Orow = O + (size_t)bh * Lsz * Dh;
  #pragma unroll
  for (int r = 0; r < 8; ++r) {
    int m = r + half * 8;
    float inv = 1.0f / rowsum[r];
    Orow[(size_t)(q0 + m) * Dh + n] = acc0[r] * inv;
    if (n < 8)
      Orow[(size_t)(q0 + m) * Dh + 16 + n] = acc1[r] * inv;
  }
}

// ------------------------------ Kernel 3 -----------------------------------
__global__ __launch_bounds__(256) void outproj_kernel(
    const float* __restrict__ O,
    const float* __restrict__ Wout_amp, const float* __restrict__ bout_amp,
    const float* __restrict__ Wout_r, const float* __restrict__ Wout_i,
    float* __restrict__ out_r, float* __restrict__ out_i,
    float* __restrict__ out_amp)
{
  int idx = blockIdx.x * blockDim.x + threadIdx.x;   // b*L + l
  if (idx >= Bsz * Lsz) return;
  int b = idx / Lsz, l = idx % Lsz;

  float oar[24], oai[24], oamp[48];
  for (int h = 0; h < Hn; ++h) {
    const float* orow = O + (((size_t)(b * Hn + h)) * Lsz + l) * Dh;
    for (int j = 0; j < 6; ++j) {
      oar[h * 6 + j] = orow[j];
      oai[h * 6 + j] = orow[6 + j];
    }
    for (int j = 0; j < 12; ++j)
      oamp[h * 12 + j] = orow[12 + j];
  }
  for (int o = 0; o < 48; ++o) {
    float a = bout_amp[o];
    for (int i = 0; i < 48; ++i) a += oamp[i] * Wout_amp[i * 48 + o];
    out_amp[(size_t)idx * 48 + o] = a;
  }
  for (int o = 0; o < 16; ++o) {
    float r = 0.f, im = 0.f;
    for (int i = 0; i < 24; ++i) {
      float wr = Wout_r[i * 16 + o], wi = Wout_i[i * 16 + o];
      r  += oar[i] * wr - oai[i] * wi;
      im += oar[i] * wi + oai[i] * wr;
    }
    out_r[(size_t)idx * 16 + o] = r;
    out_i[(size_t)idx * 16 + o] = im;
  }
}

// ------------------------------ Launcher -----------------------------------
extern "C" void kernel_launch(void* const* d_in, const int* in_sizes, int n_in,
                              void* d_out, int out_size, void* d_ws, size_t ws_size,
                              hipStream_t stream) {
  (void)in_sizes; (void)n_in; (void)out_size; (void)ws_size;

  const float* x_ang    = (const float*)d_in[0];
  const float* x_amp    = (const float*)d_in[1];
  const float* Wq_amp   = (const float*)d_in[2];
  const float* Wk_amp   = (const float*)d_in[3];
  const float* Wv_amp   = (const float*)d_in[4];
  const float* Wq_r     = (const float*)d_in[5];
  const float* Wq_i     = (const float*)d_in[6];
  const float* Wk_r     = (const float*)d_in[7];
  const float* Wk_i     = (const float*)d_in[8];
  const float* Wv_r     = (const float*)d_in[9];
  const float* Wv_i     = (const float*)d_in[10];
  const float* Wout_amp = (const float*)d_in[11];
  const float* bout_amp = (const float*)d_in[12];
  const float* Wout_r   = (const float*)d_in[13];
  const float* Wout_i   = (const float*)d_in[14];

  char* ws = (char*)d_ws;
  const size_t qBytes = (size_t)Bsz * Hn * Lsz * Dp * sizeof(_Float16); // 4 MB
  _Float16* Q  = (_Float16*)(ws);
  _Float16* K  = (_Float16*)(ws + qBytes);
  _Float16* Vt = (_Float16*)(ws + 2 * qBytes);
  float*    O  = (float*)   (ws + 3 * qBytes);                          // 6 MB

  dim3 blk(256);
  proj_kernel<<<(Bsz * Lsz) / 256, blk, 0, stream>>>(
      x_ang, x_amp, Wq_amp, Wk_amp, Wv_amp,
      Wq_r, Wq_i, Wk_r, Wk_i, Wv_r, Wv_i, Q, K, Vt);

  // 8 waves/block, 4096 waves = (b,h) x 128 q-tiles
  attn_kernel<<<(Bsz * Hn * (Lsz / 16)) / 8, blk, 0, stream>>>(Q, K, Vt, O);

  float* outr = (float*)d_out;
  float* outi = outr + (size_t)Bsz * Lsz * 16;
  float* outa = outi + (size_t)Bsz * Lsz * 16;
  outproj_kernel<<<(Bsz * Lsz) / 256, blk, 0, stream>>>(
      O, Wout_amp, bout_amp, Wout_r, Wout_i, outr, outi, outa);
}
